// SingleHeadAttention_32100585570946
// MI455X (gfx1250) — matrix operand order repaired and checked by reading, not compiled, not run on hardware
//
#include <hip/hip_runtime.h>

// ---------------------------------------------------------------------------
// Single-head attention on gfx1250 (MI455X), f16 WMMA with fp32 accumulate,
// flash-attention core with TDM (tensor_load_to_lds) k-chunk staging.
// T=4096, B=8, D=P=O=256.
// ---------------------------------------------------------------------------

#define TT 4096
#define BB 8
#define PP 256

typedef _Float16 v16h __attribute__((ext_vector_type(16)));
typedef _Float16 v8h  __attribute__((ext_vector_type(8)));
typedef float    v8f  __attribute__((ext_vector_type(8)));
typedef unsigned int v4u __attribute__((ext_vector_type(4)));
typedef int      v4i  __attribute__((ext_vector_type(4)));
typedef int      v8i  __attribute__((ext_vector_type(8)));

union H16 { v16h v; v8h h[2]; };

// D = A(16x32 f16) * B(32x16 f16) + C(16x16 f32)
__device__ __forceinline__ v8f wmma16(v16h a, v16h b, v8f c) {
    return __builtin_amdgcn_wmma_f32_16x16x32_f16(
        /*neg_a=*/false, a, /*neg_b=*/false, b,
        /*c_mod=*/(short)0, c, /*reuse_a=*/false, /*reuse_b=*/false);
}

// A-fragment (16x32, f16 source, row-major, row stride ld elements).
// lane<16 row=lane holds K {k0..k0+7, k0+16..k0+23};
// lane>=16 row=lane-16 holds K {k0+8..k0+15, k0+24..k0+31}.
__device__ __forceinline__ v16h load_a_f16(const _Float16* src, int ld, int k0) {
    int lane = threadIdx.x & 31;
    const _Float16* p = src + (lane & 15) * ld + k0 + ((lane >> 4) << 3);
    H16 u;
    u.h[0] = *(const v8h*)(p);
    u.h[1] = *(const v8h*)(p + 16);
    return u.v;
}

// Same A-fragment but fp32 source (converted on the fly).
__device__ __forceinline__ v16h load_a_f32(const float* src, int ld, int k0) {
    int lane = threadIdx.x & 31;
    const float* p = src + (lane & 15) * ld + k0 + ((lane >> 4) << 3);
    v16h r;
#pragma unroll
    for (int i = 0; i < 8; ++i) r[i] = (_Float16)p[i];
#pragma unroll
    for (int i = 0; i < 8; ++i) r[8 + i] = (_Float16)p[16 + i];
    return r;
}

// B-fragment (32x16). Source is "B transposed": row n holds column n of B,
// contiguous over K (row stride ld). lane<16: col=lane, K {k0..k0+7, k0+16..k0+23};
// lane>=16: col=lane-16, K {k0+8..k0+15, k0+24..k0+31}.
__device__ __forceinline__ v16h load_b_f16(const _Float16* src, int ld, int k0) {
    int lane = threadIdx.x & 31;
    const _Float16* p = src + (lane & 15) * ld + k0 + ((lane >> 4) << 3);
    H16 u;
    u.h[0] = *(const v8h*)(p);
    u.h[1] = *(const v8h*)(p + 16);
    return u.v;
}

// ---------------------------------------------------------------------------
// TDM: async-load a 32-row x 256-col f16 tile (row stride 256 elements in
// global) into LDS at lds_off, inserting 16B of padding after each 512B row
// so staged LDS row stride = 264 halves (bank-conflict-free b128 reads).
// Group layouts per CDNA5 ISA.
// ---------------------------------------------------------------------------
#define KSTRIDE 264   // f16 elements per staged LDS row (256 + 8 pad)

__device__ __forceinline__ void tdm_load_k(const _Float16* gsrc, unsigned lds_off) {
    unsigned long long ga = (unsigned long long)gsrc;
    v4u g0;
    g0[0] = 1u;                                   // count=1, user descriptor
    g0[1] = lds_off;                              // lds_addr (bytes)
    g0[2] = (unsigned)(ga & 0xFFFFFFFFull);       // global_addr[31:0]
    g0[3] = (unsigned)((ga >> 32) & 0x01FFFFFFull) | 0x80000000u; // [56:32]|type=2
    v8i g1;
    g1[0] = (1 << 16)      // data_size = 2 bytes
          | (1 << 20)      // pad_enable
          | (6 << 22)      // pad_interval: 128 dwords (= one 512B row)
          | (3 << 25);     // pad_amount: 4 dwords (16B)
    g1[1] = (int)(256u << 16);   // tensor_dim0[15:0]=256 in bits[63:48]
    g1[2] = (int)(4096u << 16);  // tensor_dim1[15:0]=4096 in bits[95:80]
    g1[3] = (int)(256u << 16);   // tile_dim0=256 in bits[127:112]
    g1[4] = 32;                  // tile_dim1=32 (tile_dim2=0)
    g1[5] = 256;                 // tensor_dim0_stride low 32
    g1[6] = 0;
    g1[7] = 0;
    v4i z = {0, 0, 0, 0};
#if defined(__clang_major__) && (__clang_major__ >= 23)
    v8i z8 = {0, 0, 0, 0, 0, 0, 0, 0};
    __builtin_amdgcn_tensor_load_to_lds(g0, g1, z, z, z8, 0);
#else
    __builtin_amdgcn_tensor_load_to_lds(g0, g1, z, z, 0);
#endif
}

// ---------------------------------------------------------------------------
// Kernel 0: weight prep. W_kqv [D,3P] fp32 -> WkqvT [3P,D] f16,
//           W_out [P,O] fp32 -> WoutT [O,P] f16 (B-fragment friendly).
// ---------------------------------------------------------------------------
__global__ void prep_weights(const float* __restrict__ Wkqv,
                             const float* __restrict__ Wout,
                             _Float16* __restrict__ WkqvT,
                             _Float16* __restrict__ WoutT) {
    int i = blockIdx.x * blockDim.x + threadIdx.x;
    const int N1 = 3 * PP * 256;   // 768*256
    if (i < N1) {
        int n = i >> 8, d = i & 255;
        WkqvT[n * 256 + d] = (_Float16)Wkqv[d * (3 * PP) + n];
    } else {
        int j = i - N1;
        if (j < PP * PP) {
            int n = j >> 8, p = j & 255;
            WoutT[n * 256 + p] = (_Float16)Wout[p * PP + n];
        }
    }
}

// ---------------------------------------------------------------------------
// Kernel 1: qkv = query @ W_kqv + b_kqv. 16 rows (t*B+b) per block, 8 waves;
// wave handles N-tiles nt, nt+8, ... of 48. All 8 B-fragments are loaded as
// one clause up front so WMMAs overlap outstanding loads.
// q scaled by P^-0.5 -> [B][T][P]; k -> [B][T][P]; v transposed -> [B][P][T].
// ---------------------------------------------------------------------------
__global__ __launch_bounds__(256) void qkv_kernel(
    const float* __restrict__ query, const _Float16* __restrict__ WkqvT,
    const float* __restrict__ bkqv,
    _Float16* __restrict__ qs, _Float16* __restrict__ kbuf,
    _Float16* __restrict__ vt) {
    const int r0 = blockIdx.x << 4;
    const int wave = threadIdx.x >> 5;
    const int lane = threadIdx.x & 31;
    const int hi = lane >> 4, col = lane & 15;

    v16h a[8];
#pragma unroll
    for (int kf = 0; kf < 8; ++kf)
        a[kf] = load_a_f32(query + (size_t)r0 * 256, 256, kf * 32);

    for (int nt = wave; nt < 48; nt += 8) {
        const _Float16* wb = WkqvT + (size_t)nt * 16 * 256;
        v16h bfr[8];
#pragma unroll
        for (int kf = 0; kf < 8; ++kf)
            bfr[kf] = load_b_f16(wb, 256, kf * 32);
        v8f c = {0.f, 0.f, 0.f, 0.f, 0.f, 0.f, 0.f, 0.f};
#pragma unroll
        for (int kf = 0; kf < 8; ++kf)
            c = wmma16(a[kf], bfr[kf], c);

        const int ncol = (nt << 4) + col;
        const float bias = bkqv[ncol];
        const int sect = nt >> 4;            // 0:q 1:k 2:v
        const int pcol = ncol - sect * PP;
#pragma unroll
        for (int j = 0; j < 8; ++j) {
            int r = r0 + j + (hi << 3);
            int t = r >> 3;                  // / B
            int bb = r & 7;                  // % B
            float val = c[j] + bias;
            if (sect == 0)
                qs[((size_t)bb * TT + t) * PP + pcol] = (_Float16)(val * 0.0625f);
            else if (sect == 1)
                kbuf[((size_t)bb * TT + t) * PP + pcol] = (_Float16)val;
            else
                vt[((size_t)bb * PP + pcol) * TT + t] = (_Float16)val;
        }
    }
}

// ---------------------------------------------------------------------------
// Kernel 2: flash attention. 8 waves/block, all on the SAME batch, marching
// over S in lockstep (barrier per chunk) so the 16KB k-chunk (TDM-staged in
// LDS, double-buffered) and the v-chunk (global, WGP$-hot) are shared 8 ways.
// ---------------------------------------------------------------------------
__global__ __launch_bounds__(256, 1) void attn_kernel(
    const _Float16* __restrict__ qs, const _Float16* __restrict__ kbuf,
    const _Float16* __restrict__ vt, _Float16* __restrict__ attn) {
    __shared__ __align__(16) _Float16 kstage[2][32 * KSTRIDE]; // 2 x 16.5 KB
    __shared__ __align__(16) _Float16 pstage[8][16 * 32];      // 8 KB

    const int wave = threadIdx.x >> 5;
    const int lane = threadIdx.x & 31;
    const int hi = lane >> 4, col = lane & 15;

    const int b  = blockIdx.x >> 5;                         // 32 blocks/batch
    const int t0 = (((blockIdx.x & 31) << 3) + wave) << 4;  // wave's q tile

    const _Float16* qrow  = qs   + ((size_t)b * TT + t0) * PP;
    const _Float16* kbase = kbuf + (size_t)b * TT * PP;
    const _Float16* vbase = vt   + (size_t)b * PP * TT;
    _Float16* ps = pstage[wave];

    v8f acc[16];
#pragma unroll
    for (int n = 0; n < 16; ++n)
#pragma unroll
        for (int j = 0; j < 8; ++j) acc[n][j] = 0.f;
    float mrow[8], lrow[8];
#pragma unroll
    for (int j = 0; j < 8; ++j) { mrow[j] = -1e30f; lrow[j] = 0.f; }

    // prefetch chunk 0
    if (wave == 0)
        tdm_load_k(kbase, (unsigned)(size_t)(void*)&kstage[0][0]);

    const int NCH = TT / 32;
    for (int ci = 0; ci < NCH; ++ci) {
        const int s0 = ci << 5;
        __syncthreads();   // everyone done reading the buffer we overwrite next
        if (wave == 0) {
            if (ci + 1 < NCH) {
                tdm_load_k(kbase + (size_t)(s0 + 32) * PP,
                           (unsigned)(size_t)(void*)&kstage[(ci + 1) & 1][0]);
                __builtin_amdgcn_s_wait_tensorcnt(1);  // chunk ci landed
            } else {
                __builtin_amdgcn_s_wait_tensorcnt(0);
            }
        }
        __syncthreads();   // k chunk ci visible to all waves
        const _Float16* kc = &kstage[ci & 1][0];

        // ---- scores: q(16x256) . k^T -> two 16x16 f32 tiles (k from LDS) ----
        v8f c0 = {0.f, 0.f, 0.f, 0.f, 0.f, 0.f, 0.f, 0.f};
        v8f c1 = {0.f, 0.f, 0.f, 0.f, 0.f, 0.f, 0.f, 0.f};
#pragma unroll
        for (int kf = 0; kf < 8; ++kf) {
            v16h a  = load_a_f16(qrow, PP, kf * 32);
            v16h b0 = load_b_f16(kc, KSTRIDE, kf * 32);
            v16h b1 = load_b_f16(kc + 16 * KSTRIDE, KSTRIDE, kf * 32);
            c0 = wmma16(a, b0, c0);
            c1 = wmma16(a, b1, c1);
        }
        // ---- online softmax (row = VGPR j + 8*hi, spread over 16 lanes) ----
        float scl[8];
#pragma unroll
        for (int j = 0; j < 8; ++j) {
            float mx = fmaxf(c0[j], c1[j]);
#pragma unroll
            for (int d = 8; d >= 1; d >>= 1)
                mx = fmaxf(mx, __shfl_xor(mx, d, 32));
            float mn = fmaxf(mrow[j], mx);
            float sc = __expf(mrow[j] - mn);
            float p0 = __expf(c0[j] - mn);
            float p1 = __expf(c1[j] - mn);
            float rs = p0 + p1;
#pragma unroll
            for (int d = 8; d >= 1; d >>= 1)
                rs += __shfl_xor(rs, d, 32);
            lrow[j] = lrow[j] * sc + rs;
            mrow[j] = mn;
            scl[j] = sc;
            int m = j + (hi << 3);
            ps[m * 32 + col]      = (_Float16)p0;
            ps[m * 32 + col + 16] = (_Float16)p1;
        }
#pragma unroll
        for (int n = 0; n < 16; ++n)
#pragma unroll
            for (int j = 0; j < 8; ++j) acc[n][j] *= scl[j];

        // probs C-layout -> A-fragment layout via per-wave LDS staging
        v16h pa = load_a_f16(ps, 32, 0);

        // ---- PV: probs(16x32) . v(32x256) over 16 N-tiles, v-fragments
        //      batched 4-at-a-time so loads overlap the WMMA chain ----
#pragma unroll
        for (int ng = 0; ng < 4; ++ng) {
            v16h bv[4];
#pragma unroll
            for (int i = 0; i < 4; ++i)
                bv[i] = load_b_f16(
                    vbase + (size_t)((ng * 4 + i) * 16) * TT + s0, TT, 0);
#pragma unroll
            for (int i = 0; i < 4; ++i)
                acc[ng * 4 + i] = wmma16(pa, bv[i], acc[ng * 4 + i]);
        }
    }

    // ---- epilogue: divide by row sums, store f16 [B][T][P] ----
    float rinv[8];
#pragma unroll
    for (int j = 0; j < 8; ++j) rinv[j] = 1.0f / lrow[j];
    _Float16* orow = attn + ((size_t)b * TT + t0) * PP;
#pragma unroll
    for (int n = 0; n < 16; ++n)
#pragma unroll
        for (int j = 0; j < 8; ++j)
            orow[(size_t)(j + (hi << 3)) * PP + (n << 4) + col] =
                (_Float16)(acc[n][j] * rinv[j]);
}

// ---------------------------------------------------------------------------
// Kernel 3: out = attn @ W_out + b_out -> fp32 [T,B,O]. B-fragments batched.
// ---------------------------------------------------------------------------
__global__ __launch_bounds__(256) void proj_kernel(
    const _Float16* __restrict__ attn, const _Float16* __restrict__ WoutT,
    const float* __restrict__ bout, float* __restrict__ out) {
    const int r0 = blockIdx.x << 4;
    const int wave = threadIdx.x >> 5;
    const int lane = threadIdx.x & 31;
    const int hi = lane >> 4, col = lane & 15;

    v16h a[8];
#pragma unroll
    for (int kf = 0; kf < 8; ++kf)
        a[kf] = load_a_f16(attn + (size_t)r0 * PP, PP, kf * 32);

    for (int nt = wave; nt < 16; nt += 8) {
        const _Float16* wb = WoutT + (size_t)nt * 16 * PP;
        v16h bfr[8];
#pragma unroll
        for (int kf = 0; kf < 8; ++kf)
            bfr[kf] = load_b_f16(wb, PP, kf * 32);
        v8f c = {0.f, 0.f, 0.f, 0.f, 0.f, 0.f, 0.f, 0.f};
#pragma unroll
        for (int kf = 0; kf < 8; ++kf)
            c = wmma16(a[kf], bfr[kf], c);

        const int n = (nt << 4) + col;
        const float bias = bout[n];
#pragma unroll
        for (int j = 0; j < 8; ++j) {
            int r = r0 + j + (hi << 3);
            int bb = r >> 12;               // / T
            int t  = r & (TT - 1);          // % T
            out[((size_t)t * BB + bb) * PP + n] = c[j] + bias;
        }
    }
}

// ---------------------------------------------------------------------------
// Host launcher. Workspace (~64.5 MB):
//   qs / k / vT / attn : 16 MB each (f16), then WkqvT 384KB, WoutT 128KB.
// ---------------------------------------------------------------------------
extern "C" void kernel_launch(void* const* d_in, const int* in_sizes, int n_in,
                              void* d_out, int out_size, void* d_ws, size_t ws_size,
                              hipStream_t stream) {
    const float* query = (const float*)d_in[0];
    const float* Wkqv  = (const float*)d_in[1];
    const float* bkqv  = (const float*)d_in[2];
    const float* Wout  = (const float*)d_in[3];
    const float* bout  = (const float*)d_in[4];
    float* out = (float*)d_out;

    const size_t QKV_BYTES = (size_t)BB * TT * PP * sizeof(_Float16); // 16 MB
    char* w = (char*)d_ws;
    _Float16* qs    = (_Float16*)(w);
    _Float16* kbuf  = (_Float16*)(w + QKV_BYTES);
    _Float16* vt    = (_Float16*)(w + 2 * QKV_BYTES);
    _Float16* attn  = (_Float16*)(w + 3 * QKV_BYTES);
    _Float16* WkqvT = (_Float16*)(w + 4 * QKV_BYTES);
    _Float16* WoutT = (_Float16*)(w + 4 * QKV_BYTES + (size_t)3 * PP * 256 * 2);

    {
        int total = 3 * PP * 256 + PP * PP;
        prep_weights<<<(total + 255) / 256, 256, 0, stream>>>(Wkqv, Wout, WkqvT, WoutT);
    }
    qkv_kernel<<<(TT * BB) / 16, 256, 0, stream>>>(query, WkqvT, bkqv, qs, kbuf, vt);
    // 8 q-tiles per block sharing k/v chunks; 256 blocks total
    attn_kernel<<<(BB * (TT / 16)) / 8, 256, 0, stream>>>(qs, kbuf, vt, attn);
    proj_kernel<<<(TT * BB) / 16, 256, 0, stream>>>(attn, WoutT, bout, out);
}
